// PointNet2Classification_39986145525967
// MI455X (gfx1250) — matrix operand order repaired and checked by reading, not compiled, not run on hardware
//
#include <hip/hip_runtime.h>
#include <hip/hip_bf16.h>

typedef _Float16 h16;
typedef __attribute__((ext_vector_type(16))) _Float16 v16h;
typedef __attribute__((ext_vector_type(8)))  _Float16 v8h;
typedef __attribute__((ext_vector_type(8)))  float    v8f;

#define BATCH 16
#define NPTS  32768

// ---------------------------------------------------------------------------
// Weight conversion: f32 (cout, cin) -> f16 (coutP, cinP) with zero padding.
// ---------------------------------------------------------------------------
__global__ void cvt_w_kernel(const float* __restrict__ src, h16* __restrict__ dst,
                             int cout, int cin, int coutP, int cinP) {
  int tid = blockIdx.x * blockDim.x + threadIdx.x;
  int total = coutP * cinP;
  if (tid >= total) return;
  int o = tid / cinP, i = tid % cinP;
  float v = (o < cout && i < cin) ? src[(size_t)o * cin + i] : 0.0f;
  dst[tid] = (h16)v;
}

// ---------------------------------------------------------------------------
// Farthest point sampling. One block per batch; mind[] lives in global scratch
// (L2-resident), argmax via LDS tree reduction. Matches reference: sidx[0]=0,
// ties pick lowest index.
// ---------------------------------------------------------------------------
__global__ void fps_kernel(const float* __restrict__ pts, int stride, int n, int npoint,
                           float* __restrict__ mind, int* __restrict__ sidx) {
  int b = blockIdx.x;
  const float* P = pts + (size_t)b * n * stride;
  float* md = mind + (size_t)b * n;
  int t = threadIdx.x, nt = blockDim.x;

  __shared__ float sv[256];
  __shared__ int   si[256];
  __shared__ float lp[3];

  for (int i = t; i < n; i += nt) md[i] = 1e10f;
  __syncthreads();

  for (int s = 0; s < npoint; s++) {
    if (t == 0) {
      int last = (s == 0) ? 0 : si[0];
      sidx[b * npoint + s] = last;
      const float* q = P + (size_t)last * stride;
      lp[0] = q[0]; lp[1] = q[1]; lp[2] = q[2];
    }
    __syncthreads();
    float lx = lp[0], ly = lp[1], lz = lp[2];
    float bestV = -1.0f; int bestI = 0;
    for (int i = t; i < n; i += nt) {
      const float* q = P + (size_t)i * stride;
      float dx = q[0] - lx, dy = q[1] - ly, dz = q[2] - lz;
      float d = dx * dx + dy * dy + dz * dz;
      float m = md[i];
      m = fminf(m, d);
      md[i] = m;
      if (m > bestV) { bestV = m; bestI = i; }
    }
    sv[t] = bestV; si[t] = bestI;
    __syncthreads();
    for (int off = nt >> 1; off > 0; off >>= 1) {
      if (t < off) {
        if (sv[t + off] > sv[t] || (sv[t + off] == sv[t] && si[t + off] < si[t])) {
          sv[t] = sv[t + off]; si[t] = si[t + off];
        }
      }
      __syncthreads();
    }
  }
}

// ---------------------------------------------------------------------------
// Gather sampled centroid coordinates.
// ---------------------------------------------------------------------------
__global__ void gather_xyz_kernel(const float* __restrict__ pts, int stride, int n,
                                  const int* __restrict__ sidx, int S,
                                  float* __restrict__ nx) {
  int tid = blockIdx.x * blockDim.x + threadIdx.x;
  if (tid >= BATCH * S) return;
  int b = tid / S;
  int pi = sidx[tid];
  const float* p = pts + ((size_t)b * n + pi) * stride;
  nx[tid * 3 + 0] = p[0];
  nx[tid * 3 + 1] = p[1];
  nx[tid * 3 + 2] = p[2];
}

// ---------------------------------------------------------------------------
// Ball query: one wave32 per centroid. Emits first nsample indices within
// radius in ascending order (ballot + prefix-popc), pads with first hit (0 if
// no hits) -- matches the reference exactly.
// ---------------------------------------------------------------------------
__global__ void ball_query_kernel(const float* __restrict__ pts, int stride, int n,
                                  const float* __restrict__ ctr, int S, float r2,
                                  int nsample, int* __restrict__ gidx) {
  int wpb = blockDim.x >> 5;
  int wid = blockIdx.x * wpb + (threadIdx.x >> 5);
  int lane = threadIdx.x & 31;
  if (wid >= BATCH * S) return;
  int b = wid / S;
  const float* P = pts + (size_t)b * n * stride;
  float cx = ctr[wid * 3 + 0], cy = ctr[wid * 3 + 1], cz = ctr[wid * 3 + 2];
  int* out = gidx + (size_t)wid * nsample;

  int cnt = 0, first = 0, any = 0;
  for (int base = 0; base < n && cnt < nsample; base += 32) {
    int i = base + lane;
    bool hit = false;
    if (i < n) {
      const float* q = P + (size_t)i * stride;
      float dx = q[0] - cx, dy = q[1] - cy, dz = q[2] - cz;
      hit = (dx * dx + dy * dy + dz * dz) < r2;
    }
    unsigned mask = __builtin_amdgcn_ballot_w32(hit);
    if (!any && mask) { first = base + __builtin_ctz(mask); any = 1; }
    if (hit) {
      int pos = cnt + __builtin_popcount(mask & ((1u << lane) - 1u));
      if (pos < nsample) out[pos] = i;
    }
    cnt += __builtin_popcount(mask);
  }
  if (cnt > nsample) cnt = nsample;
  int pad = any ? first : 0;
  for (int j = cnt + lane; j < nsample; j += 32) out[j] = pad;
}

// ---------------------------------------------------------------------------
// Grouping kernels: build f16 GEMM inputs (K padded to multiples of 32).
// ---------------------------------------------------------------------------
__global__ void group1_kernel(const float* __restrict__ pc, const float* __restrict__ ctr,
                              const int* __restrict__ gidx, h16* __restrict__ out) {
  int tid = blockIdx.x * blockDim.x + threadIdx.x;
  const int total = BATCH * 128 * 64;
  if (tid >= total) return;
  int b = tid / (128 * 64);
  int s = (tid / 64) % 128;
  int pi = gidx[tid];
  const float* p = pc + ((size_t)b * NPTS + pi) * 6;
  const float* c = ctr + ((size_t)b * 128 + s) * 3;
  h16* o = out + (size_t)tid * 32;
  o[0] = (h16)(p[0] - c[0]);
  o[1] = (h16)(p[1] - c[1]);
  o[2] = (h16)(p[2] - c[2]);
  o[3] = (h16)p[3]; o[4] = (h16)p[4]; o[5] = (h16)p[5];
#pragma unroll
  for (int k = 6; k < 32; k++) o[k] = (h16)0.0f;
}

__global__ void group2_kernel(const float* __restrict__ xyz1, const h16* __restrict__ feat1,
                              const float* __restrict__ ctr, const int* __restrict__ gidx,
                              h16* __restrict__ out) {
  int tid = blockIdx.x * blockDim.x + threadIdx.x;
  const int total = BATCH * 32 * 64;
  if (tid >= total) return;
  int b = tid / (32 * 64);
  int s = (tid / 64) % 32;
  int pi = gidx[tid];
  const float* p = xyz1 + ((size_t)b * 128 + pi) * 3;
  const h16*   f = feat1 + ((size_t)b * 128 + pi) * 128;
  const float* c = ctr + ((size_t)b * 32 + s) * 3;
  h16* o = out + (size_t)tid * 160;
  o[0] = (h16)(p[0] - c[0]);
  o[1] = (h16)(p[1] - c[1]);
  o[2] = (h16)(p[2] - c[2]);
  for (int k = 0; k < 128; k++) o[3 + k] = f[k];
#pragma unroll
  for (int k = 131; k < 160; k++) o[k] = (h16)0.0f;
}

__global__ void group_all_kernel(const float* __restrict__ xyz2, const h16* __restrict__ feat2,
                                 h16* __restrict__ out) {
  int tid = blockIdx.x * blockDim.x + threadIdx.x;
  if (tid >= BATCH * 32) return;
  const float* p = xyz2 + (size_t)tid * 3;
  const h16*   f = feat2 + (size_t)tid * 256;
  h16* o = out + (size_t)tid * 288;
  o[0] = (h16)p[0]; o[1] = (h16)p[1]; o[2] = (h16)p[2];
  for (int k = 0; k < 256; k++) o[3 + k] = f[k];
#pragma unroll
  for (int k = 259; k < 288; k++) o[k] = (h16)0.0f;
}

// ---------------------------------------------------------------------------
// WMMA linear layer: D(M,Nc) = relu(A(M,K) * W(Nc,K)^T + bias).
// One wave32 per 16x16 output tile; K stepped by 32 with
// v_wmma_f32_16x16x32_f16 (f32 accumulation).
// A operand layout (16-bit A 16x32): lanes 0-15 hold K k..k+15 (contiguous, same order as B),
// lanes 16-31 hold K k+16..k+31 for row M = lane&15.
// B operand: lane n holds W[col][k + sel*16 .. +15] (contiguous, coalesced).
// C/D: acc[r] -> (M = r + 8*sel, N = lane&15) of the tile.
// ---------------------------------------------------------------------------
__global__ void wmma_linear_kernel(const h16* __restrict__ A, const h16* __restrict__ W,
                                   const float* __restrict__ bias, int nBias,
                                   h16* __restrict__ D, int K, int Nc, int doRelu) {
  int lane = threadIdx.x & 31;
  int sel  = lane >> 4;
  int l15  = lane & 15;
  int tm = blockIdx.x, tn = blockIdx.y;
  int row = tm * 16 + l15;
  int col = tn * 16 + l15;
  const h16* arow = A + (size_t)row * K;
  const h16* wrow = W + (size_t)col * K;

  v8f acc = {};
  for (int k = 0; k < K; k += 32) {
    v8h alo = *(const v8h*)(arow + k + sel * 16);
    v8h ahi = *(const v8h*)(arow + k + sel * 16 + 8);
    v16h a;
#pragma unroll
    for (int i = 0; i < 8; i++) { a[i] = alo[i]; a[i + 8] = ahi[i]; }
    v16h bm = *(const v16h*)(wrow + k + sel * 16);
    acc = __builtin_amdgcn_wmma_f32_16x16x32_f16(
        /*neg_a=*/false, a, /*neg_b=*/false, bm,
        /*c_mod=*/(short)0, acc, /*reuse_a=*/false, /*reuse_b=*/false);
  }

  float bv = (col < nBias) ? bias[col] : 0.0f;
  h16* drow = D + (size_t)(tm * 16 + sel * 8) * Nc + col;
#pragma unroll
  for (int r = 0; r < 8; r++) {
    float v = acc[r] + bv;
    if (doRelu) v = fmaxf(v, 0.0f);
    drow[(size_t)r * Nc] = (h16)v;
  }
}

// ---------------------------------------------------------------------------
// Max-pool over the neighborhood dim: in(rows*ns, C) -> out(rows, C).
// ---------------------------------------------------------------------------
__global__ void maxpool_kernel(const h16* __restrict__ in, h16* __restrict__ out,
                               int rows, int ns, int C) {
  int tid = blockIdx.x * blockDim.x + threadIdx.x;
  int total = rows * C;
  if (tid >= total) return;
  int r = tid / C, c = tid % C;
  const h16* p = in + (size_t)r * ns * C + c;
  float m = -1e30f;
  for (int j = 0; j < ns; j++) m = fmaxf(m, (float)p[(size_t)j * C]);
  out[tid] = (h16)m;
}

// ---------------------------------------------------------------------------
// Final output assembly: objectness (16) then semantics (16x8), f32.
// ---------------------------------------------------------------------------
__global__ void epilogue_kernel(const h16* __restrict__ cls, const h16* __restrict__ sem,
                                float* __restrict__ out) {
  int t = threadIdx.x;
  if (t < 16) {
    out[t] = (float)cls[t * 16 + 0];
  } else if (t < 144) {
    int i = t - 16;
    out[16 + i] = (float)sem[(i / 8) * 16 + (i % 8)];
  }
}

// ---------------------------------------------------------------------------
extern "C" void kernel_launch(void* const* d_in, const int* in_sizes, int n_in,
                              void* d_out, int out_size, void* d_ws, size_t ws_size,
                              hipStream_t stream) {
  (void)in_sizes; (void)n_in; (void)out_size; (void)ws_size;
  const float* pc = (const float*)d_in[0];
  char* ws = (char*)d_ws;

  size_t off = 0;
  auto alloc = [&](size_t bytes) -> size_t {
    size_t o = off;
    off = (o + bytes + 255) & ~(size_t)255;
    return o;
  };

  // Layer table: {W input index, cout, cin, coutPad, cinPad}
  struct LD { int wIdx; int cout, cin, coutP, cinP; };
  const LD L[15] = {
      {1, 128, 6, 128, 32},   {3, 128, 128, 128, 128}, {5, 128, 128, 128, 128},   // sa1
      {7, 128, 131, 128, 160},{9, 128, 128, 128, 128}, {11, 256, 128, 256, 128},  // sa2
      {13, 256, 259, 256, 288},{15, 256, 256, 256, 256},{17, 512, 256, 512, 256}, // sa3
      {19, 256, 512, 256, 512},{21, 256, 256, 256, 256},{23, 1, 256, 16, 256},    // cls
      {25, 256, 512, 256, 512},{27, 256, 256, 256, 256},{29, 8, 256, 16, 256},    // sem
  };
  size_t wOff[15];
  for (int l = 0; l < 15; l++) wOff[l] = alloc((size_t)L[l].coutP * L[l].cinP * 2);

  size_t oMind  = alloc((size_t)BATCH * NPTS * 4);
  size_t oSidx1 = alloc((size_t)BATCH * 128 * 4);
  size_t oNx1   = alloc((size_t)BATCH * 128 * 3 * 4);
  size_t oGidx1 = alloc((size_t)BATCH * 128 * 64 * 4);
  size_t oSidx2 = alloc((size_t)BATCH * 32 * 4);
  size_t oNx2   = alloc((size_t)BATCH * 32 * 3 * 4);
  size_t oGidx2 = alloc((size_t)BATCH * 32 * 64 * 4);
  size_t oFeat1 = alloc((size_t)BATCH * 128 * 128 * 2);
  size_t oFeat2 = alloc((size_t)BATCH * 32 * 256 * 2);
  size_t oF3    = alloc((size_t)BATCH * 512 * 2);
  size_t oH1    = alloc((size_t)16 * 256 * 2);
  size_t oH2    = alloc((size_t)16 * 256 * 2);
  size_t oCls   = alloc((size_t)16 * 16 * 2);
  size_t oSem   = alloc((size_t)16 * 16 * 2);
  size_t oBufG  = alloc((size_t)131072 * 32 * 2);   // 8 MB
  size_t oBufA  = alloc((size_t)131072 * 128 * 2);  // 32 MB
  size_t oBufB  = alloc((size_t)131072 * 128 * 2);  // 32 MB

  float* mind  = (float*)(ws + oMind);
  int*   sidx1 = (int*)(ws + oSidx1);
  float* nx1   = (float*)(ws + oNx1);
  int*   gidx1 = (int*)(ws + oGidx1);
  int*   sidx2 = (int*)(ws + oSidx2);
  float* nx2   = (float*)(ws + oNx2);
  int*   gidx2 = (int*)(ws + oGidx2);
  h16*   feat1 = (h16*)(ws + oFeat1);
  h16*   feat2 = (h16*)(ws + oFeat2);
  h16*   f3    = (h16*)(ws + oF3);
  h16*   bufG  = (h16*)(ws + oBufG);
  h16*   bufA  = (h16*)(ws + oBufA);
  h16*   bufB  = (h16*)(ws + oBufB);
  auto wh = [&](int l) { return (h16*)(ws + wOff[l]); };
  auto bf = [&](int l) { return (const float*)d_in[L[l].wIdx + 1]; };

  // --- Convert all weights to padded f16 ---
  for (int l = 0; l < 15; l++) {
    int total = L[l].coutP * L[l].cinP;
    cvt_w_kernel<<<(total + 255) / 256, 256, 0, stream>>>(
        (const float*)d_in[L[l].wIdx], wh(l), L[l].cout, L[l].cin, L[l].coutP, L[l].cinP);
  }

  // --- SA1: FPS(128) + ball query(0.2, 64) + MLP [32->128->128->128] + pool ---
  fps_kernel<<<BATCH, 256, 0, stream>>>(pc, 6, NPTS, 128, mind, sidx1);
  gather_xyz_kernel<<<(BATCH * 128 + 255) / 256, 256, 0, stream>>>(pc, 6, NPTS, sidx1, 128, nx1);
  ball_query_kernel<<<(BATCH * 128) / 4, 128, 0, stream>>>(pc, 6, NPTS, nx1, 128, 0.04f, 64, gidx1);
  group1_kernel<<<(BATCH * 128 * 64 + 255) / 256, 256, 0, stream>>>(pc, nx1, gidx1, bufG);
  wmma_linear_kernel<<<dim3(8192, 8),  32, 0, stream>>>(bufG, wh(0), bf(0), 128, bufA, 32, 128, 1);
  wmma_linear_kernel<<<dim3(8192, 8),  32, 0, stream>>>(bufA, wh(1), bf(1), 128, bufB, 128, 128, 1);
  wmma_linear_kernel<<<dim3(8192, 8),  32, 0, stream>>>(bufB, wh(2), bf(2), 128, bufA, 128, 128, 1);
  maxpool_kernel<<<(2048 * 128 + 255) / 256, 256, 0, stream>>>(bufA, feat1, 2048, 64, 128);

  // --- SA2: FPS(32) + ball query(0.4, 64) + MLP [160->128->128->256] + pool ---
  fps_kernel<<<BATCH, 128, 0, stream>>>(nx1, 3, 128, 32, mind, sidx2);
  gather_xyz_kernel<<<2, 256, 0, stream>>>(nx1, 3, 128, sidx2, 32, nx2);
  ball_query_kernel<<<(BATCH * 32) / 4, 128, 0, stream>>>(nx1, 3, 128, nx2, 32, 0.16f, 64, gidx2);
  group2_kernel<<<(BATCH * 32 * 64 + 255) / 256, 256, 0, stream>>>(nx1, feat1, nx2, gidx2, bufB);
  wmma_linear_kernel<<<dim3(2048, 8),  32, 0, stream>>>(bufB, wh(3), bf(3), 128, bufA, 160, 128, 1);
  wmma_linear_kernel<<<dim3(2048, 8),  32, 0, stream>>>(bufA, wh(4), bf(4), 128, bufG, 128, 128, 1);
  wmma_linear_kernel<<<dim3(2048, 16), 32, 0, stream>>>(bufG, wh(5), bf(5), 256, bufA, 128, 256, 1);
  maxpool_kernel<<<(512 * 256 + 255) / 256, 256, 0, stream>>>(bufA, feat2, 512, 64, 256);

  // --- SA3 (group all): MLP [288->256->256->512] + pool over 32 ---
  group_all_kernel<<<2, 256, 0, stream>>>(nx2, feat2, bufB);
  wmma_linear_kernel<<<dim3(32, 16), 32, 0, stream>>>(bufB, wh(6), bf(6), 256, bufG, 288, 256, 1);
  wmma_linear_kernel<<<dim3(32, 16), 32, 0, stream>>>(bufG, wh(7), bf(7), 256, bufA, 256, 256, 1);
  wmma_linear_kernel<<<dim3(32, 32), 32, 0, stream>>>(bufA, wh(8), bf(8), 512, bufG, 256, 512, 1);
  maxpool_kernel<<<(16 * 512 + 255) / 256, 256, 0, stream>>>(bufG, f3, 16, 32, 512);

  // --- Heads (M=16, one tile row each) ---
  h16* h1 = (h16*)(ws + oH1);
  h16* h2 = (h16*)(ws + oH2);
  h16* clsO = (h16*)(ws + oCls);
  h16* semO = (h16*)(ws + oSem);
  // cls: 512 -> 256 -> 256 -> 1 (padded to 16)
  wmma_linear_kernel<<<dim3(1, 16), 32, 0, stream>>>(f3, wh(9),  bf(9),  256, h1, 512, 256, 1);
  wmma_linear_kernel<<<dim3(1, 16), 32, 0, stream>>>(h1, wh(10), bf(10), 256, h2, 256, 256, 1);
  wmma_linear_kernel<<<dim3(1, 1),  32, 0, stream>>>(h2, wh(11), bf(11), 1,   clsO, 256, 16, 0);
  // sem: 512 -> 256 -> 256 -> 8 (padded to 16)
  wmma_linear_kernel<<<dim3(1, 16), 32, 0, stream>>>(f3, wh(12), bf(12), 256, h1, 512, 256, 1);
  wmma_linear_kernel<<<dim3(1, 16), 32, 0, stream>>>(h1, wh(13), bf(13), 256, h2, 256, 256, 1);
  wmma_linear_kernel<<<dim3(1, 1),  32, 0, stream>>>(h2, wh(14), bf(14), 8,   semO, 256, 16, 0);

  epilogue_kernel<<<1, 144, 0, stream>>>(clsO, semO, (float*)d_out);
}
